// TemporalAttention_8211977469970
// MI455X (gfx1250) — matrix operand order repaired and checked by reading, not compiled, not run on hardware
//
#include <hip/hip_runtime.h>
#include <hip/hip_bf16.h>
#include <stdint.h>

// ---------------------------------------------------------------------------
// Types for CDNA5 WMMA (wave32): D(16x16 f32) = A(16x32 bf16) x B(32x16 bf16) + C
// ---------------------------------------------------------------------------
typedef __attribute__((ext_vector_type(16))) __bf16 bf16x16;
typedef __attribute__((ext_vector_type(8)))  float  f32x8;

union BFrag {
    bf16x16 v;      // 32 bytes = 8 VGPRs
    uint4   q[2];   // two 16B chunks for b128 loads
};

__device__ __forceinline__ unsigned short f2bf(float f) {
    unsigned int u = __builtin_bit_cast(unsigned int, f);
    u += 0x7FFFu + ((u >> 16) & 1u);      // round-to-nearest-even
    return (unsigned short)(u >> 16);
}

#define NB   512
#define NSEQ 64
#define NC   768
#define NH   12

// ---------------------------------------------------------------------------
// Pack qkv_w [3C, C] f32 -> per-fragment bf16 blocks.
// Fragment id = ((h*3 + m)*4 + ntile)*24 + kstep ; layout [frag][lane][16].
// B(32x16) layout: lanes 0-15 hold K=0..15 (cols N=lane), lanes 16-31 K=16..31.
// ---------------------------------------------------------------------------
__global__ void k_pack_qkv(const float* __restrict__ w, unsigned short* __restrict__ wp) {
    int p = blockIdx.x * blockDim.x + threadIdx.x;          // 1,769,472 total
    if (p >= 12 * 3 * 4 * 24 * 512) return;
    int i     = p & 15;
    int lane  = (p >> 4) & 31;
    int kstep = (p >> 9) % 24;
    int rest  = p / (512 * 24);
    int ntile = rest & 3;
    int m     = (rest >> 2) % 3;
    int h     = rest / 12;
    int row = m * NC + h * 64 + ntile * 16 + (lane & 15);    // output channel (N)
    int col = kstep * 32 + ((lane < 16) ? i : 16 + i);       // K
    wp[p] = f2bf(w[row * NC + col]);
}

// Pack proj_w [C, C]: frag id = ntile_global(0..47)*24 + kstep
__global__ void k_pack_proj(const float* __restrict__ w, unsigned short* __restrict__ wp) {
    int p = blockIdx.x * blockDim.x + threadIdx.x;          // 589,824 total
    if (p >= 48 * 24 * 512) return;
    int i     = p & 15;
    int lane  = (p >> 4) & 31;
    int kstep = (p >> 9) % 24;
    int ntg   = p / (512 * 24);
    int row = ntg * 16 + (lane & 15);
    int col = kstep * 32 + ((lane < 16) ? i : 16 + i);
    wp[p] = f2bf(w[row * NC + col]);
}

// ---------------------------------------------------------------------------
// Fused QKV-projection + attention per (b, h).  192 threads = 6 waves.
// ---------------------------------------------------------------------------
__global__ __launch_bounds__(192) void k_qkv_attn(
    const float* __restrict__ x,            // [B, N, C] f32
    const unsigned short* __restrict__ wq,  // packed qkv fragments (bf16)
    const float* __restrict__ qkv_b,        // [3C]
    unsigned short* __restrict__ hb)        // [B*N, C] bf16 head-major output
{
    __shared__ unsigned short xs[64 * 128];   // 16 KB: x chunk, bf16
    __shared__ unsigned short qs[64 * 64];    //  8 KB: q (later reused as p)
    __shared__ unsigned short ksm[64 * 64];   //  8 KB: k row-major
    __shared__ unsigned short vt[64 * 64];    //  8 KB: v transposed [d][j]
    __shared__ float          ss[64 * 64];    // 16 KB: attention scores f32

    const int tid   = threadIdx.x;
    const int wave  = tid >> 5;
    const int lane  = tid & 31;
    const int lrow  = lane & 15;
    const int lhalf = lane >> 4;              // 0: lanes 0-15, 1: lanes 16-31
    const int b     = blockIdx.x / NH;
    const int h     = blockIdx.x % NH;

    // ---- Phase 1: q,k,v = x[b] @ Wqkv_h^T  (each wave: 2 strips x 4 mtiles) ----
    const int s0 = wave * 2;                  // strips s0, s0+1 of 12 (m*4+ntile)
    f32x8 acc[2][4];
#pragma unroll
    for (int s = 0; s < 2; ++s)
#pragma unroll
        for (int mt = 0; mt < 4; ++mt)
#pragma unroll
            for (int e = 0; e < 8; ++e) acc[s][mt][e] = 0.f;

    const float* xrow = x + (size_t)(b * NSEQ) * NC;

    for (int kc = 0; kc < 6; ++kc) {          // K chunks of 128
        // stage 64x128 f32 -> bf16 LDS
        for (int idx = tid; idx < 2048; idx += 192) {   // 2048 float4
            int r  = idx >> 5;
            int c4 = (idx & 31) * 4;
            float4 f = *(const float4*)&xrow[r * NC + kc * 128 + c4];
            uint2 pk;
            pk.x = (unsigned)f2bf(f.x) | ((unsigned)f2bf(f.y) << 16);
            pk.y = (unsigned)f2bf(f.z) | ((unsigned)f2bf(f.w) << 16);
            *(uint2*)&xs[r * 128 + c4] = pk;
        }
        __syncthreads();

#pragma unroll
        for (int kk = 0; kk < 4; ++kk) {      // 4 k-steps of 32 per chunk
            const int kstep = kc * 4 + kk;
            const int cb = kk * 32 + lhalf * 16;
            BFrag a[4];
#pragma unroll
            for (int mt = 0; mt < 4; ++mt) {  // A frags shared across strips
                int row = mt * 16 + lrow;
                a[mt].q[0] = *(const uint4*)&xs[row * 128 + cb];
                a[mt].q[1] = *(const uint4*)&xs[row * 128 + cb + 8];
            }
#pragma unroll
            for (int s = 0; s < 2; ++s) {
                const int strip = s0 + s;
                const int m  = strip >> 2;
                const int nt = strip & 3;
                const int fragid = ((h * 3 + m) * 4 + nt) * 24 + kstep;
                BFrag bf;
                const uint4* bp = (const uint4*)&wq[(size_t)fragid * 512 + lane * 16];
                bf.q[0] = bp[0];
                bf.q[1] = bp[1];
#pragma unroll
                for (int mt = 0; mt < 4; ++mt)
                    acc[s][mt] = __builtin_amdgcn_wmma_f32_16x16x32_bf16(
                        false, a[mt].v, false, bf.v, (short)0, acc[s][mt], false, false);
            }
        }
        __syncthreads();
    }

    // ---- Phase 2: +bias, q,k row-major / v transposed into LDS (bf16) ----
#pragma unroll
    for (int s = 0; s < 2; ++s) {
        const int strip = s0 + s;
        const int m  = strip >> 2;
        const int nt = strip & 3;
        const float bias = qkv_b[m * NC + h * 64 + nt * 16 + lrow];
#pragma unroll
        for (int mt = 0; mt < 4; ++mt)
#pragma unroll
            for (int r = 0; r < 8; ++r) {
                int M = mt * 16 + r + lhalf * 8;         // C/D layout row
                int Ncol = nt * 16 + lrow;               // C/D layout col
                unsigned short bv = f2bf(acc[s][mt][r] + bias);
                if (m == 0)      qs[M * 64 + Ncol] = bv;
                else if (m == 1) ksm[M * 64 + Ncol] = bv;
                else             vt[Ncol * 64 + M] = bv; // v^T: [d][j]
            }
    }
    __syncthreads();

    // ---- Phase 3: S = (q @ k^T) * hd^-0.5 (mask is a faithful no-op) ----
    for (int t = wave; t < 16; t += 6) {
        const int mt = t >> 2, nt = t & 3;
        f32x8 sacc;
#pragma unroll
        for (int e = 0; e < 8; ++e) sacc[e] = 0.f;
#pragma unroll
        for (int kstep = 0; kstep < 2; ++kstep) {
            BFrag a, bf;
            int row = mt * 16 + lrow;
            int cb  = kstep * 32 + lhalf * 16;
            a.q[0] = *(const uint4*)&qs[row * 64 + cb];
            a.q[1] = *(const uint4*)&qs[row * 64 + cb + 8];
            int j   = nt * 16 + lrow;                    // key index (col of S)
            int off = j * 64 + kstep * 32 + lhalf * 16;  // contiguous d-range
            bf.q[0] = *(const uint4*)&ksm[off];
            bf.q[1] = *(const uint4*)&ksm[off + 8];
            sacc = __builtin_amdgcn_wmma_f32_16x16x32_bf16(
                false, a.v, false, bf.v, (short)0, sacc, false, false);
        }
#pragma unroll
        for (int r = 0; r < 8; ++r)
            ss[(mt * 16 + r + lhalf * 8) * 64 + nt * 16 + lrow] = sacc[r] * 0.125f;
    }
    __syncthreads();

    // ---- Phase 4: row softmax; write p (bf16) into qs ----
    if (tid < 64) {
        float* srow = &ss[tid * 64];
        float mx = -3.0e38f;
        for (int j = 0; j < 64; ++j) mx = fmaxf(mx, srow[j]);
        float sum = 0.f;
        for (int j = 0; j < 64; ++j) {
            float e = __expf(srow[j] - mx);
            srow[j] = e;
            sum += e;
        }
        float inv = 1.f / sum;
        for (int j = 0; j < 64; ++j) qs[tid * 64 + j] = f2bf(srow[j] * inv);
    }
    __syncthreads();

    // ---- Phase 5: O = P @ V ; store bf16 into hb[b,n, h*64+d] ----
    unsigned short* hrow = hb + (size_t)(b * NSEQ) * NC + h * 64;
    for (int t = wave; t < 16; t += 6) {
        const int mt = t >> 2, nt = t & 3;
        f32x8 oacc;
#pragma unroll
        for (int e = 0; e < 8; ++e) oacc[e] = 0.f;
#pragma unroll
        for (int kstep = 0; kstep < 2; ++kstep) {
            BFrag a, bf;
            int row = mt * 16 + lrow;
            int cb  = kstep * 32 + lhalf * 16;
            a.q[0] = *(const uint4*)&qs[row * 64 + cb];
            a.q[1] = *(const uint4*)&qs[row * 64 + cb + 8];
            int d   = nt * 16 + lrow;                    // head dim (col of O)
            int off = d * 64 + kstep * 32 + lhalf * 16;  // contiguous j-range in v^T
            bf.q[0] = *(const uint4*)&vt[off];
            bf.q[1] = *(const uint4*)&vt[off + 8];
            oacc = __builtin_amdgcn_wmma_f32_16x16x32_bf16(
                false, a.v, false, bf.v, (short)0, oacc, false, false);
        }
#pragma unroll
        for (int r = 0; r < 8; ++r) {
            int n = mt * 16 + r + lhalf * 8;
            hrow[(size_t)n * NC + nt * 16 + lrow] = f2bf(oacc[r]);
        }
    }
}

// ---------------------------------------------------------------------------
// Output projection: out[32768,768] = hb @ proj_w^T + proj_b.  256 thr = 8 waves,
// each block one 64x64 tile, each wave 2 of 16 WMMA tiles, K staged in LDS.
// ---------------------------------------------------------------------------
__global__ __launch_bounds__(256) void k_proj(
    const unsigned short* __restrict__ hb,
    const unsigned short* __restrict__ wp,
    const float* __restrict__ pb,
    float* __restrict__ out)
{
    __shared__ unsigned short as_[64 * 128];  // 16 KB

    const int tid   = threadIdx.x;
    const int wave  = tid >> 5;
    const int lane  = tid & 31;
    const int lrow  = lane & 15;
    const int lhalf = lane >> 4;
    const int mblk  = blockIdx.x / 12;
    const int nblk  = blockIdx.x % 12;
    const int t0    = wave * 2;

    const unsigned short* arow = hb + (size_t)(mblk * 64) * NC;

    f32x8 acc[2];
#pragma unroll
    for (int ti = 0; ti < 2; ++ti)
#pragma unroll
        for (int e = 0; e < 8; ++e) acc[ti][e] = 0.f;

    for (int kc = 0; kc < 6; ++kc) {
        for (int idx = tid; idx < 1024; idx += 256) {    // 1024 uint4
            int r  = idx >> 4;
            int c8 = (idx & 15) * 8;
            *(uint4*)&as_[r * 128 + c8] = *(const uint4*)&arow[r * NC + kc * 128 + c8];
        }
        __syncthreads();
#pragma unroll
        for (int kk = 0; kk < 4; ++kk) {
            const int kstep = kc * 4 + kk;
            const int cb = kk * 32 + lhalf * 16;
#pragma unroll
            for (int ti = 0; ti < 2; ++ti) {
                const int t  = t0 + ti;
                const int nt = t >> 2;
                const int mt = t & 3;
                BFrag a, bf;
                int row = mt * 16 + lrow;
                a.q[0] = *(const uint4*)&as_[row * 128 + cb];
                a.q[1] = *(const uint4*)&as_[row * 128 + cb + 8];
                const int fragid = (nblk * 4 + nt) * 24 + kstep;
                const uint4* bp = (const uint4*)&wp[(size_t)fragid * 512 + lane * 16];
                bf.q[0] = bp[0];
                bf.q[1] = bp[1];
                acc[ti] = __builtin_amdgcn_wmma_f32_16x16x32_bf16(
                    false, a.v, false, bf.v, (short)0, acc[ti], false, false);
            }
        }
        __syncthreads();
    }

#pragma unroll
    for (int ti = 0; ti < 2; ++ti) {
        const int t  = t0 + ti;
        const int nt = t >> 2;
        const int mt = t & 3;
        const int col = nblk * 64 + nt * 16 + lrow;
        const float bias = pb[col];
#pragma unroll
        for (int r = 0; r < 8; ++r) {
            int row = mblk * 64 + mt * 16 + r + lhalf * 8;
            out[(size_t)row * NC + col] = acc[ti][r] + bias;
        }
    }
}

// ---------------------------------------------------------------------------
extern "C" void kernel_launch(void* const* d_in, const int* in_sizes, int n_in,
                              void* d_out, int out_size, void* d_ws, size_t ws_size,
                              hipStream_t stream) {
    const float* x      = (const float*)d_in[0];   // [512,64,768]
    const float* qkv_w  = (const float*)d_in[1];   // [2304,768]
    const float* qkv_b  = (const float*)d_in[2];   // [2304]
    const float* proj_w = (const float*)d_in[3];   // [768,768]
    const float* proj_b = (const float*)d_in[4];   // [768]
    float* out = (float*)d_out;

    // Workspace: head-out bf16 (48 MB) + packed qkv (3.4 MB) + packed proj (1.1 MB)
    unsigned short* hb = (unsigned short*)d_ws;
    unsigned short* wq = hb + (size_t)32768 * 768;
    unsigned short* wp = wq + (size_t)2304 * 768;

    k_pack_qkv <<<6912, 256, 0, stream>>>(qkv_w, wq);
    k_pack_proj<<<2304, 256, 0, stream>>>(proj_w, wp);
    k_qkv_attn <<<NB * NH, 192, 0, stream>>>(x, wq, qkv_b, hb);
    k_proj     <<<NB * NH, 256, 0, stream>>>(hb, wp, proj_b, out);
}
